// M3Care_37546604101821
// MI455X (gfx1250) — matrix operand order repaired and checked by reading, not compiled, not run on hardware
//
#include <hip/hip_runtime.h>

typedef __attribute__((ext_vector_type(16))) _Float16 v16h;
typedef __attribute__((ext_vector_type(4)))  _Float16 h4;
typedef __attribute__((ext_vector_type(8)))  float    v8f;

// ---------------- problem constants ----------------
constexpr int N_  = 2048;
constexpr int T_  = 32;
constexpr int D_  = 256;
constexpr int NE_ = 34;                 // T + 2 tokens
constexpr int ZROWS = N_ * NE_;         // 69632
constexpr size_t ND = (size_t)N_ * D_;          // 524288
constexpr size_t NN = (size_t)N_ * N_;          // 4194304
constexpr size_t ZN = (size_t)ZROWS * D_;       // 17825792

// ---------------- workspace layout (float offsets) ----------------
constexpr size_t SC    = 0;        // 64 scalars: 0 bwp,1 bwe,2 ssq_p,3 ssq_e,4 lstab
constexpr size_t MEANV = 64;       // 256 mean + 256 var
constexpr size_t PART  = 576;      // 1024 reduction partials
constexpr size_t IMP0  = 8192;     // imputed[3] : 3*ND
constexpr size_t ZB    = IMP0 + 3 * ND;
constexpr size_t TB1   = ZB  + ZN;
constexpr size_t TB2   = TB1 + ZN;
constexpr size_t TB3   = TB2 + ZN;
constexpr size_t TFB   = TB3 + ZN;   // 69632*1024 FFN hidden
// early-phase aliases (dead before transformer GEMMs start)
constexpr size_t E_BN  = TB1;
constexpr size_t P_BN  = TB1 + ND;
constexpr size_t H1B   = TB1 + 2 * ND;
constexpr size_t H2B   = TB1 + 3 * ND;
constexpr size_t P_RAW = TB1 + 4 * ND;
constexpr size_t SQP   = TB1 + 5 * ND;
constexpr size_t SQE   = SQP + N_;
constexpr size_t GPM   = TB2;
constexpr size_t GEM   = TB2 + NN;
constexpr size_t SIMM  = TB2 + 2 * NN;
constexpr size_t AGG   = TB2 + 3 * NN;
constexpr size_t GX1   = TB3;
constexpr size_t GH1   = TB3 + ND;
constexpr size_t GX2   = TB3 + 2 * ND;
constexpr size_t GH2   = TB3 + 3 * ND;

constexpr int FLAG_BIAS = 1;
constexpr int FLAG_RELU = 2;
constexpr int RED_BLOCKS = 1024;

// ---------------- WMMA GEMM: C = op(A[M,K] @ B + bias) ----------------
// block = 256 threads (8 waves); block tile 128x128, K-step 32.
// Register double-buffering: tile k+1's global loads are issued before the
// compute phase of tile k, so VMEM latency overlaps the 8 WMMAs per K-step.
#define TM 128
#define TN 128
#define TK 32
__global__ __launch_bounds__(256)
void gemm_wmma(const float* __restrict__ A, const float* __restrict__ B,
               const float* __restrict__ bias, float* __restrict__ C,
               int M, int N, int K, int lda, int ldb, int ldc,
               int transB, int flags)
{
    __shared__ _Float16 As[TM][TK];     // 8 KB
    __shared__ _Float16 Bs[TN][TK];     // 8 KB, stored as [n][k]
    const int tid  = threadIdx.x;
    const int wave = tid >> 5;          // 0..7
    const int lane = tid & 31;
    const int bm = blockIdx.y * TM;
    const int bn = blockIdx.x * TN;

    // staging coordinates (fixed per thread)
    const int ar  = (tid + 0 * 256) >> 3;        // pattern base; full set below
    (void)ar;

    auto loadA = [&](int k0, float4* r) {
        #pragma unroll
        for (int i = 0; i < 4; ++i) {
            int f = tid + i * 256;               // 0..1023 float4 index
            int rr = f >> 3, c4 = (f & 7) * 4;   // 8 float4 per 32-col row
            r[i] = *reinterpret_cast<const float4*>(
                &A[(size_t)(bm + rr) * lda + k0 + c4]);
        }
    };
    auto loadB = [&](int k0, float4* r) {
        if (transB) {
            #pragma unroll
            for (int i = 0; i < 4; ++i) {
                int f = tid + i * 256;
                int rr = f >> 3, c4 = (f & 7) * 4;
                r[i] = *reinterpret_cast<const float4*>(
                    &B[(size_t)(bn + rr) * ldb + k0 + c4]);
            }
        } else {
            #pragma unroll
            for (int i = 0; i < 4; ++i) {
                int f = tid + i * 256;               // 32 k-rows x 32 float4
                int kk = f >> 5, n4 = (f & 31) * 4;  // 32 float4 per 128-col row
                r[i] = *reinterpret_cast<const float4*>(
                    &B[(size_t)(k0 + kk) * ldb + bn + n4]);
            }
        }
    };
    auto storeA = [&](const float4* r) {
        #pragma unroll
        for (int i = 0; i < 4; ++i) {
            int f = tid + i * 256;
            int rr = f >> 3, c4 = (f & 7) * 4;
            h4 hv = { (_Float16)r[i].x, (_Float16)r[i].y,
                      (_Float16)r[i].z, (_Float16)r[i].w };
            *reinterpret_cast<h4*>(&As[rr][c4]) = hv;
        }
    };
    auto storeB = [&](const float4* r) {
        if (transB) {
            #pragma unroll
            for (int i = 0; i < 4; ++i) {
                int f = tid + i * 256;
                int rr = f >> 3, c4 = (f & 7) * 4;
                h4 hv = { (_Float16)r[i].x, (_Float16)r[i].y,
                          (_Float16)r[i].z, (_Float16)r[i].w };
                *reinterpret_cast<h4*>(&Bs[rr][c4]) = hv;
            }
        } else {
            #pragma unroll
            for (int i = 0; i < 4; ++i) {
                int f = tid + i * 256;
                int kk = f >> 5, n4 = (f & 31) * 4;
                Bs[n4 + 0][kk] = (_Float16)r[i].x;
                Bs[n4 + 1][kk] = (_Float16)r[i].y;
                Bs[n4 + 2][kk] = (_Float16)r[i].z;
                Bs[n4 + 3][kk] = (_Float16)r[i].w;
            }
        }
    };

    v8f acc[8] = {};
    float4 ra[4], rb[4];

    loadA(0, ra);
    loadB(0, rb);

    for (int k0 = 0; k0 < K; k0 += TK) {
        // convert + store the tile already in registers
        storeA(ra);
        storeB(rb);
        __syncthreads();

        // issue next tile's global loads; waits land in next iteration's store
        if (k0 + TK < K) {
            loadA(k0 + TK, ra);
            loadB(k0 + TK, rb);
        }

        const int row  = lane & 15;
        const int kbA0 = (lane >> 4) * 8;         // A frag: K 0..7 / 8..15 (+16)
        const int kbB  = (lane >> 4) * 8;         // B frag: K 0..7 / 8..15 (+16)
        v16h afrag;
        const _Float16* ap = &As[wave * 16 + row][0];
        #pragma unroll
        for (int j = 0; j < 8; ++j) {
            afrag[j]     = ap[kbA0 + j];
            afrag[8 + j] = ap[kbA0 + 16 + j];
        }
        #pragma unroll
        for (int t = 0; t < 8; ++t) {
            const _Float16* bp = &Bs[t * 16 + row][0];
            v16h bfrag;
            #pragma unroll
            for (int j = 0; j < 8; ++j) { bfrag[j] = bp[kbB + j]; bfrag[8 + j] = bp[kbB + 16 + j]; }
            acc[t] = __builtin_amdgcn_wmma_f32_16x16x32_f16(
                false, afrag, false, bfrag, (short)0, acc[t], false, false);
        }
        __syncthreads();
    }

    // store per documented C/D layout
    const int rbase = bm + wave * 16 + (lane >> 4) * 8;
    const int cin = lane & 15;
    #pragma unroll
    for (int t = 0; t < 8; ++t) {
        int col = bn + t * 16 + cin;
        #pragma unroll
        for (int r = 0; r < 8; ++r) {
            float v = acc[t][r];
            if (flags & FLAG_BIAS) v += bias[col];
            if (flags & FLAG_RELU) v = fmaxf(v, 0.f);
            C[(size_t)(rbase + r) * ldc + col] = v;
        }
    }
}
#undef TM
#undef TN
#undef TK

// ---------------- helpers ----------------
__device__ __forceinline__ float blockReduceSum256(float v, float* sh) {
    int tid = threadIdx.x;
    sh[tid] = v; __syncthreads();
    for (int s = 128; s > 0; s >>= 1) {
        if (tid < s) sh[tid] += sh[tid + s];
        __syncthreads();
    }
    float r = sh[0]; __syncthreads();
    return r;
}
__device__ __forceinline__ float sigm(float x) { return 1.f / (1.f + __expf(-x)); }

__global__ void init_scalars(float* scal) { if (threadIdx.x < 64) scal[threadIdx.x] = 0.f; }

// per-column batch stats over N rows
__global__ __launch_bounds__(256)
void colstats_kernel(const float* __restrict__ X, long rowstride, int relu_in,
                     float* __restrict__ mean, float* __restrict__ var)
{
    __shared__ float sh[256];
    int c = blockIdx.x;
    float s = 0.f, s2 = 0.f;
    for (int r = threadIdx.x; r < N_; r += 256) {
        float v = X[(size_t)r * rowstride + c];
        if (relu_in) v = fmaxf(v, 0.f);
        s += v; s2 += v * v;
    }
    float S  = blockReduceSum256(s, sh);
    float S2 = blockReduceSum256(s2, sh);
    if (threadIdx.x == 0) {
        float m = S / N_;
        mean[c] = m;
        var[c]  = fmaxf(S2 / N_ - m * m, 0.f);
    }
}

__global__ __launch_bounds__(256)
void bn_apply_kernel(const float* __restrict__ X, long rowstride, int relu_in,
                     const float* __restrict__ mean, const float* __restrict__ var,
                     const float* __restrict__ g, const float* __restrict__ b,
                     float* __restrict__ Y)
{
    for (size_t idx = (size_t)blockIdx.x * 256 + threadIdx.x; idx < ND;
         idx += (size_t)gridDim.x * 256) {
        size_t r = idx / D_; int c = idx % D_;
        float v = X[r * rowstride + c];
        if (relu_in) v = fmaxf(v, 0.f);
        Y[idx] = (v - mean[c]) * rsqrtf(var[c] + 1e-5f) * g[c] + b[c];
    }
}

__global__ __launch_bounds__(256)
void row_sumsq_kernel(const float* __restrict__ X, float* __restrict__ sq)
{
    __shared__ float sh[256];
    int r = blockIdx.x;
    float v = X[(size_t)r * D_ + threadIdx.x];
    float S = blockReduceSum256(v * v, sh);
    if (threadIdx.x == 0) sq[r] = S;
}

__global__ __launch_bounds__(256)
void reduce_sumsq_partial(const float* __restrict__ X, long rows, int cols,
                          long rowstride, float* __restrict__ partials)
{
    __shared__ float sh[256];
    size_t total = (size_t)rows * cols;
    float acc = 0.f;
    for (size_t idx = (size_t)blockIdx.x * 256 + threadIdx.x; idx < total;
         idx += (size_t)RED_BLOCKS * 256) {
        size_t r = idx / cols; int c = (int)(idx % cols);
        float v = X[r * rowstride + c];
        acc += v * v;
    }
    float S = blockReduceSum256(acc, sh);
    if (threadIdx.x == 0) partials[blockIdx.x] = S;
}

__global__ __launch_bounds__(256)
void reduce_l2_partial(const float* __restrict__ G, const float* __restrict__ sq,
                       float* __restrict__ partials)
{
    __shared__ float sh[256];
    float acc = 0.f;
    for (size_t idx = (size_t)blockIdx.x * 256 + threadIdx.x; idx < NN;
         idx += (size_t)RED_BLOCKS * 256) {
        int i = (int)(idx >> 11), j = (int)(idx & 2047);
        acc += fmaxf(sq[i] + sq[j] - 2.f * G[idx], 0.f);
    }
    float S = blockReduceSum256(acc, sh);
    if (threadIdx.x == 0) partials[blockIdx.x] = S;
}

__global__ __launch_bounds__(256)
void reduce_finish(const float* __restrict__ partials, float* __restrict__ outslot)
{
    __shared__ float sh[256];
    float acc = 0.f;
    for (int i = threadIdx.x; i < RED_BLOCKS; i += 256) acc += partials[i];
    float S = blockReduceSum256(acc, sh);
    if (threadIdx.x == 0) *outslot = S;
}

__global__ void lstab_acc_kernel(float* scal)
{
    if (threadIdx.x == 0)
        scal[4] += fabsf(sqrtf(scal[2]) - sqrtf(scal[3]));
}

// fused multi-bandwidth gaussian-kernel similarity accumulation
__global__ __launch_bounds__(256)
void gram_post_kernel(const float* __restrict__ Gp, const float* __restrict__ Ge,
                      const float* __restrict__ sqp, const float* __restrict__ sqe,
                      const float* __restrict__ scal, const float* __restrict__ simi_eps,
                      int m, const int* __restrict__ mk, int mstride,
                      float* __restrict__ sim, int accum)
{
    const float norm = 1.f / ((float)((size_t)N_ * N_ - N_) * 2.f); // /(n^2-n)/GK_MUL^(GK_NUM//2)
    float bwp = scal[0] * norm;
    float bwe = scal[1] * norm;
    float ip0 = 1.f / bwp, ip1 = 0.5f * ip0, ip2 = 0.25f * ip0;
    float ie0 = 1.f / bwe, ie1 = 0.5f * ie0, ie2 = 0.25f * ie0;
    float eps = sigm(simi_eps[m]);
    for (size_t idx = (size_t)blockIdx.x * 256 + threadIdx.x; idx < NN;
         idx += (size_t)gridDim.x * 256) {
        int i = (int)(idx >> 11), j = (int)(idx & 2047);
        float l2p = fmaxf(sqp[i] + sqp[j] - 2.f * Gp[idx], 0.f);
        float l2e = fmaxf(sqe[i] + sqe[j] - 2.f * Ge[idx], 0.f);
        float wgk = __expf(-l2p * ip0) + __expf(-l2p * ip1) + __expf(-l2p * ip2);
        float gk  = __expf(-l2e * ie0) + __expf(-l2e * ie1) + __expf(-l2e * ie2);
        float mm  = (mk[(size_t)i * mstride] != 0 && mk[(size_t)j * mstride] != 0) ? 1.f : 0.f;
        float val = ((1.f - eps) * wgk + eps) * gk * mm;
        sim[idx] = accum ? (sim[idx] + val) : val;
    }
}

__global__ __launch_bounds__(256)
void agg_kernel(const float* __restrict__ sim, const int* __restrict__ mk0,
                const int* __restrict__ mk1, const int* __restrict__ mk2,
                const float* __restrict__ dissim, float* __restrict__ agg)
{
    float thr = sigm(dissim[0]);
    for (size_t idx = (size_t)blockIdx.x * 256 + threadIdx.x; idx < NN;
         idx += (size_t)gridDim.x * 256) {
        int i = (int)(idx >> 11), j = (int)(idx & 2047);
        float cnt = 0.f;
        cnt += (mk0[(size_t)i * T_] != 0 && mk0[(size_t)j * T_] != 0) ? 1.f : 0.f;
        cnt += (mk1[i] != 0 && mk1[j] != 0) ? 1.f : 0.f;
        cnt += (mk2[i] != 0 && mk2[j] != 0) ? 1.f : 0.f;
        float a = sim[idx] / cnt;
        agg[idx] = (a > thr) ? a : 0.f;
    }
}

__global__ __launch_bounds__(256)
void impute_kernel(const float* __restrict__ e, long estride,
                   const float* __restrict__ h, const float* __restrict__ aw,
                   const float* __restrict__ ab, const int* __restrict__ mk,
                   int mstride, float* __restrict__ out)
{
    __shared__ float sh[256];
    int n = blockIdx.x, c = threadIdx.x;
    float ev = e[(size_t)n * estride + c];
    float hv = h[(size_t)n * D_ + c];
    float w  = aw[c];
    float de = blockReduceSum256(ev * w, sh);
    float dh = blockReduceSum256(hv * w, sh);
    float s = sigm(de + ab[0]);
    float o = sigm(dh + ab[0]);
    s = s / (s + o);
    float imp = s * ev + (1.f - s) * hv;
    out[(size_t)n * D_ + c] = (mk[(size_t)n * mstride] != 0) ? imp : hv;
}

__global__ __launch_bounds__(256)
void assemble_z_kernel(const float* __restrict__ emb0, const float* __restrict__ imp0,
                       const float* __restrict__ imp1, const float* __restrict__ imp2,
                       const float* __restrict__ type_emb, float* __restrict__ z)
{
    const float LOG1E4_D = 9.210340372f / (float)D_;
    size_t total = (size_t)N_ * NE_ * D_;
    for (size_t idx = (size_t)blockIdx.x * 256 + threadIdx.x; idx < total;
         idx += (size_t)gridDim.x * 256) {
        size_t n = idx / ((size_t)NE_ * D_);
        int rem = (int)(idx % ((size_t)NE_ * D_));
        int l = rem / D_, d = rem % D_;
        float v;
        if (l < T_) {
            float base = (l == 0) ? imp0[n * D_ + d]
                                  : emb0[n * (size_t)T_ * D_ + (size_t)l * D_ + d];
            int j = d >> 1;
            float ang = (float)l * __expf(-(float)(2 * j) * LOG1E4_D);
            float pe = (d & 1) ? __cosf(ang) : __sinf(ang);
            v = base + type_emb[d] + pe;
        } else if (l == T_) {
            v = imp1[n * D_ + d] + type_emb[D_ + d];
        } else {
            v = imp2[n * D_ + d] + type_emb[2 * D_ + d];
        }
        z[idx] = v;
    }
}

// masked multi-head attention for one (batch, head); writes O over Q buffer.
__global__ __launch_bounds__(256)
void attn_kernel(const float* __restrict__ q, const float* __restrict__ k,
                 const float* __restrict__ v, float* __restrict__ o,
                 const int* __restrict__ msk0, const int* __restrict__ msk2,
                 int heads, int dh, float scale)
{
    __shared__ float s[NE_][NE_ + 2];
    int b = blockIdx.x;
    int n = b / heads, h = b % heads;
    int tid = threadIdx.x;
    auto tok = [&](int l) -> bool {
        if (l < T_) return msk0[(size_t)n * T_ + l] != 0;
        if (l == T_) return true;
        return msk2[n] != 0;
    };
    for (int idx = tid; idx < NE_ * NE_; idx += 256) {
        int l = idx / NE_, m = idx % NE_;
        float val = -1e9f;
        if (tok(l) && tok(m)) {
            const float* qp = q + ((size_t)(n * NE_ + l)) * D_ + h * dh;
            const float* kp = k + ((size_t)(n * NE_ + m)) * D_ + h * dh;
            float acc = 0.f;
            for (int d = 0; d < dh; ++d) acc += qp[d] * kp[d];
            val = acc * scale;
        }
        s[l][m] = val;
    }
    __syncthreads();
    if (tid < NE_) {
        int l = tid;
        float mx = -3.4e38f;
        for (int m = 0; m < NE_; ++m) mx = fmaxf(mx, s[l][m]);
        float sum = 0.f;
        for (int m = 0; m < NE_; ++m) { float e = __expf(s[l][m] - mx); s[l][m] = e; sum += e; }
        float inv = 1.f / sum;
        for (int m = 0; m < NE_; ++m) s[l][m] *= inv;
    }
    __syncthreads();
    for (int idx = tid; idx < NE_ * dh; idx += 256) {
        int l = idx / dh, d = idx % dh;
        float acc = 0.f;
        for (int m = 0; m < NE_; ++m)
            acc += s[l][m] * v[((size_t)(n * NE_ + m)) * D_ + h * dh + d];
        o[((size_t)(n * NE_ + l)) * D_ + h * dh + d] = acc;
    }
}

// out = LN(X + R) * g + b  (optional relu)
__global__ __launch_bounds__(256)
void ln_kernel(const float* __restrict__ X, const float* __restrict__ R,
               const float* __restrict__ g, const float* __restrict__ b,
               float* __restrict__ out, int relu_out)
{
    __shared__ float sh[256];
    size_t row = blockIdx.x;
    int c = threadIdx.x;
    float x = X[row * D_ + c] + R[row * D_ + c];
    float mean = blockReduceSum256(x, sh) * (1.f / D_);
    float d0 = x - mean;
    float var = blockReduceSum256(d0 * d0, sh) * (1.f / D_);
    float y = d0 * rsqrtf(var + 1e-5f) * g[c] + b[c];
    if (relu_out) y = fmaxf(y, 0.f);
    out[row * D_ + c] = y;
}

__global__ __launch_bounds__(256)
void final_head_kernel(const float* __restrict__ Y, const float* __restrict__ w2,
                       const float* __restrict__ b2, const float* __restrict__ scal,
                       float* __restrict__ out)
{
    __shared__ float sh[256];
    int i = blockIdx.x, tid = threadIdx.x;
    float acc = Y[(size_t)i * 512 + tid] * w2[tid]
              + Y[(size_t)i * 512 + 256 + tid] * w2[256 + tid];
    float S = blockReduceSum256(acc, sh);
    if (tid == 0) out[i] = S + b2[0];
    if (i == 0 && tid == 0) out[N_] = scal[4];   // lstab
}

// ---------------- host ----------------
extern "C" void kernel_launch(void* const* d_in, const int* in_sizes, int n_in,
                              void* d_out, int out_size, void* d_ws, size_t ws_size,
                              hipStream_t stream)
{
    (void)in_sizes; (void)n_in; (void)out_size; (void)ws_size;
    const float* emb0     = (const float*)d_in[0];
    const float* emb1     = (const float*)d_in[1];
    const float* emb2     = (const float*)d_in[2];
    const int*   msk0     = (const int*)d_in[3];
    const int*   msk1     = (const int*)d_in[4];
    const int*   msk2     = (const int*)d_in[5];
    const float* simi_w   = (const float*)d_in[6];
    const float* simi_b   = (const float*)d_in[7];
    const float* bn_g     = (const float*)d_in[8];
    const float* bn_b     = (const float*)d_in[9];
    const float* simi_eps = (const float*)d_in[10];
    const float* dissim   = (const float*)d_in[11];
    const float* gcn_w    = (const float*)d_in[12];
    const float* gcn_b    = (const float*)d_in[13];
    const float* adapt_w  = (const float*)d_in[14];
    const float* adapt_b  = (const float*)d_in[15];
    const float* type_emb = (const float*)d_in[16];
    const float* attn_w   = (const float*)d_in[17];
    const float* attn_b   = (const float*)d_in[18];
    const float* ffn_w1   = (const float*)d_in[19];
    const float* ffn_b1   = (const float*)d_in[20];
    const float* ffn_w2   = (const float*)d_in[21];
    const float* ffn_b2   = (const float*)d_in[22];
    const float* ln_g     = (const float*)d_in[23];
    const float* ln_b     = (const float*)d_in[24];
    const float* out_w1   = (const float*)d_in[25];
    const float* out_b1   = (const float*)d_in[26];
    const float* out_w2   = (const float*)d_in[27];
    const float* out_b2   = (const float*)d_in[28];

    float* ws  = (float*)d_ws;
    float* out = (float*)d_out;
    float* scal  = ws + SC;
    float* meanv = ws + MEANV;
    float* part  = ws + PART;

    auto gemm = [&](const float* A, const float* B, const float* bias, float* C,
                    int M, int Nc, int K, int lda, int ldb, int ldc,
                    int transB, int flags) {
        dim3 g(Nc / 128, M / 128);
        gemm_wmma<<<g, 256, 0, stream>>>(A, B, bias, C, M, Nc, K, lda, ldb, ldc, transB, flags);
    };

    hipLaunchKernelGGL(init_scalars, dim3(1), dim3(64), 0, stream, scal);

    const size_t DD = (size_t)D_ * D_;

    // ---------- Phase A: per-modality similarity ----------
    for (int m = 0; m < 3; ++m) {
        const float* src = (m == 0) ? emb0 : (m == 1) ? emb1 : emb2;
        long estride = (m == 0) ? (long)T_ * D_ : (long)D_;
        const int* mk = (m == 0) ? msk0 : (m == 1) ? msk1 : msk2;
        int mstride = (m == 0) ? T_ : 1;

        // MLP3: relu, relu, linear
        gemm(src,        simi_w + ((size_t)m * 3 + 0) * DD, simi_b + ((size_t)m * 3 + 0) * D_,
             ws + H1B,   N_, D_, D_, (int)estride, D_, D_, 0, FLAG_BIAS | FLAG_RELU);
        gemm(ws + H1B,   simi_w + ((size_t)m * 3 + 1) * DD, simi_b + ((size_t)m * 3 + 1) * D_,
             ws + H2B,   N_, D_, D_, D_, D_, D_, 0, FLAG_BIAS | FLAG_RELU);
        gemm(ws + H2B,   simi_w + ((size_t)m * 3 + 2) * DD, simi_b + ((size_t)m * 3 + 2) * D_,
             ws + P_RAW, N_, D_, D_, D_, D_, D_, 0, FLAG_BIAS);

        // lstab: |‖p‖F − ‖e‖F|
        reduce_sumsq_partial<<<RED_BLOCKS, 256, 0, stream>>>(ws + P_RAW, N_, D_, D_, part);
        reduce_finish<<<1, 256, 0, stream>>>(part, scal + 2);
        reduce_sumsq_partial<<<RED_BLOCKS, 256, 0, stream>>>(src, N_, D_, estride, part);
        reduce_finish<<<1, 256, 0, stream>>>(part, scal + 3);
        lstab_acc_kernel<<<1, 1, 0, stream>>>(scal);

        // BN(relu(p)) and BN(e)
        colstats_kernel<<<D_, 256, 0, stream>>>(ws + P_RAW, D_, 1, meanv, meanv + D_);
        bn_apply_kernel<<<2048, 256, 0, stream>>>(ws + P_RAW, D_, 1, meanv, meanv + D_,
                                                  bn_g, bn_b, ws + P_BN);
        colstats_kernel<<<D_, 256, 0, stream>>>(src, estride, 0, meanv, meanv + D_);
        bn_apply_kernel<<<2048, 256, 0, stream>>>(src, estride, 0, meanv, meanv + D_,
                                                  bn_g, bn_b, ws + E_BN);

        // Gram matrices x @ x^T via WMMA (transB)
        gemm(ws + P_BN, ws + P_BN, nullptr, ws + GPM, N_, N_, D_, D_, D_, N_, 1, 0);
        gemm(ws + E_BN, ws + E_BN, nullptr, ws + GEM, N_, N_, D_, D_, D_, N_, 1, 0);
        row_sumsq_kernel<<<N_, 256, 0, stream>>>(ws + P_BN, ws + SQP);
        row_sumsq_kernel<<<N_, 256, 0, stream>>>(ws + E_BN, ws + SQE);

        // bandwidth sums
        reduce_l2_partial<<<RED_BLOCKS, 256, 0, stream>>>(ws + GPM, ws + SQP, part);
        reduce_finish<<<1, 256, 0, stream>>>(part, scal + 0);
        reduce_l2_partial<<<RED_BLOCKS, 256, 0, stream>>>(ws + GEM, ws + SQE, part);
        reduce_finish<<<1, 256, 0, stream>>>(part, scal + 1);

        gram_post_kernel<<<4096, 256, 0, stream>>>(ws + GPM, ws + GEM, ws + SQP, ws + SQE,
                                                   scal, simi_eps, m, mk, mstride,
                                                   ws + SIMM, (m > 0) ? 1 : 0);
    }

    // ---------- agg ----------
    agg_kernel<<<4096, 256, 0, stream>>>(ws + SIMM, msk0, msk1, msk2, dissim, ws + AGG);

    // ---------- GCN + imputation ----------
    for (int m = 0; m < 3; ++m) {
        const float* src = (m == 0) ? emb0 : (m == 1) ? emb1 : emb2;
        long estride = (m == 0) ? (long)T_ * D_ : (long)D_;
        const int* mk = (m == 0) ? msk0 : (m == 1) ? msk1 : msk2;
        int mstride = (m == 0) ? T_ : 1;

        gemm(src,      gcn_w + ((size_t)m * 2 + 0) * DD, nullptr, ws + GX1,
             N_, D_, D_, (int)estride, D_, D_, 0, 0);
        gemm(ws + AGG, ws + GX1, gcn_b + ((size_t)m * 2 + 0) * D_, ws + GH1,
             N_, D_, N_, N_, D_, D_, 0, FLAG_BIAS | FLAG_RELU);
        gemm(ws + GH1, gcn_w + ((size_t)m * 2 + 1) * DD, nullptr, ws + GX2,
             N_, D_, D_, D_, D_, D_, 0, 0);
        gemm(ws + AGG, ws + GX2, gcn_b + ((size_t)m * 2 + 1) * D_, ws + GH2,
             N_, D_, N_, N_, D_, D_, 0, FLAG_BIAS | FLAG_RELU);
        impute_kernel<<<N_, 256, 0, stream>>>(src, estride, ws + GH2,
                                              adapt_w + (size_t)m * D_, adapt_b + m,
                                              mk, mstride, ws + IMP0 + (size_t)m * ND);
    }

    // ---------- assemble tokens ----------
    assemble_z_kernel<<<4096, 256, 0, stream>>>(emb0, ws + IMP0, ws + IMP0 + ND,
                                                ws + IMP0 + 2 * ND, type_emb, ws + ZB);

    // ---------- transformer (2 layers) ----------
    float* z  = ws + ZB;
    float* t1 = ws + TB1;
    float* t2 = ws + TB2;
    float* t3 = ws + TB3;
    float* tf = ws + TFB;
    for (int li = 0; li < 2; ++li) {
        int heads = (li == 0) ? 4 : 1;
        int dh = D_ / heads;
        float scale = rsqrtf((float)dh);
        const float* Wq = attn_w + ((size_t)li * 4 + 0) * DD;
        const float* Wk = attn_w + ((size_t)li * 4 + 1) * DD;
        const float* Wv = attn_w + ((size_t)li * 4 + 2) * DD;
        const float* Wo = attn_w + ((size_t)li * 4 + 3) * DD;
        const float* bq = attn_b + ((size_t)li * 4 + 0) * D_;
        const float* bk = attn_b + ((size_t)li * 4 + 1) * D_;
        const float* bv = attn_b + ((size_t)li * 4 + 2) * D_;
        const float* bo = attn_b + ((size_t)li * 4 + 3) * D_;

        gemm(z, Wq, bq, t1, ZROWS, D_, D_, D_, D_, D_, 0, FLAG_BIAS);
        gemm(z, Wk, bk, t2, ZROWS, D_, D_, D_, D_, D_, 0, FLAG_BIAS);
        gemm(z, Wv, bv, t3, ZROWS, D_, D_, D_, D_, D_, 0, FLAG_BIAS);
        attn_kernel<<<N_ * heads, 256, 0, stream>>>(t1, t2, t3, t1, msk0, msk2,
                                                    heads, dh, scale);
        gemm(t1, Wo, bo, t2, ZROWS, D_, D_, D_, D_, D_, 0, FLAG_BIAS);
        ln_kernel<<<ZROWS, 256, 0, stream>>>(z, t2, ln_g + ((size_t)li * 2 + 0) * D_,
                                             ln_b + ((size_t)li * 2 + 0) * D_, t1, 0);
        gemm(t1, ffn_w1 + (size_t)li * D_ * 1024, ffn_b1 + (size_t)li * 1024, tf,
             ZROWS, 1024, D_, D_, 1024, 1024, 0, FLAG_BIAS | FLAG_RELU);
        gemm(tf, ffn_w2 + (size_t)li * 1024 * D_, ffn_b2 + (size_t)li * D_, t3,
             ZROWS, D_, 1024, 1024, D_, D_, 0, FLAG_BIAS);
        ln_kernel<<<ZROWS, 256, 0, stream>>>(t1, t3, ln_g + ((size_t)li * 2 + 1) * D_,
                                             ln_b + ((size_t)li * 2 + 1) * D_, z, 1);
    }

    // ---------- output head ----------
    gemm(z, out_w1, out_b1, t1, N_, 512, NE_ * D_, NE_ * D_, 512, 512, 0,
         FLAG_BIAS | FLAG_RELU);
    final_head_kernel<<<N_, 256, 0, stream>>>(t1, out_w2, out_b2, scal, out);
}
